// SRPBlock_89060441850065
// MI455X (gfx1250) — matrix operand order repaired and checked by reading, not compiled, not run on hardware
//
#include <hip/hip_runtime.h>
#include <hip/hip_bf16.h>
#include <stddef.h>
#include <stdint.h>

typedef _Float16 half_t;
typedef __attribute__((ext_vector_type(16))) _Float16 v16h;
typedef __attribute__((ext_vector_type(8)))  _Float16 v8h;
typedef __attribute__((ext_vector_type(2)))  _Float16 h2v;
typedef __attribute__((ext_vector_type(8)))  float    v8f;

#define HH 256
#define WW 256
#define BB 4
#define COUT 64
#define HWSZ (HH*WW)

// ---------------------------------------------------------------------------
// Weight pre-pack: w[cout][cin_real][3][3] (f32) -> A-fragment layout (f16)
// wf[t][kb][mb][lane][e]: 16 halves per lane = exact WMMA A-matrix image.
// A 16-bit 16x32 (this code's k order; B uses the same): m = lane%16, hi = lane/16,
// element e -> K = hi*16 + e.
// ---------------------------------------------------------------------------
__global__ void pack_wfrag_k(const float* __restrict__ w, half_t* __restrict__ wf,
                             int cin_real, int KB) {
  int total = 9 * KB * 4 * 32 * 16;
  int i = blockIdx.x * 256 + threadIdx.x;
  if (i >= total) return;
  int e    = i & 15;
  int lane = (i >> 4) & 31;
  int mb   = (i >> 9) & 3;
  int kb   = (i >> 11) % KB;
  int t    = (i >> 11) / KB;
  int m  = lane & 15;
  int hi = lane >> 4;
  int k  = hi * 16 + e;
  int cin  = kb * 32 + k;
  int cout = mb * 16 + m;
  float v = 0.f;
  if (cin < cin_real) v = w[((size_t)cout * cin_real + cin) * 9 + t];
  wf[i] = (half_t)v;
}

// ---------------------------------------------------------------------------
// Pack x = concat(old,new) (f32 NCHW) into f16 NHWC [B][H][W][64]
// ---------------------------------------------------------------------------
__global__ void pack_x_k(const float* __restrict__ o, const float* __restrict__ n,
                         half_t* __restrict__ x) {
  size_t i = (size_t)blockIdx.x * 256 + threadIdx.x;   // < B*HW*64, NHWC linear
  int c = (int)(i & 63);
  size_t pix = (i >> 6) % HWSZ;
  size_t b   = (i >> 6) / HWSZ;
  float v = (c < 32) ? o[(b * 32 + c) * HWSZ + pix]
                     : n[(b * 32 + (c - 32)) * HWSZ + pix];
  x[i] = (half_t)v;
}

__global__ void mul_k(const half_t* __restrict__ a, const half_t* __restrict__ b,
                      half_t* __restrict__ o) {
  size_t i = (size_t)blockIdx.x * 256 + threadIdx.x;
  o[i] = (half_t)((float)a[i] * (float)b[i]);
}

// out (f32, NCHW) = t16 + x16 (both f16 NHWC)
__global__ void final_k(const half_t* __restrict__ t, const half_t* __restrict__ x,
                        float* __restrict__ o) {
  size_t i = (size_t)blockIdx.x * 256 + threadIdx.x;   // NCHW linear (coalesced store)
  size_t pix = i % HWSZ;
  int    c   = (int)((i / HWSZ) & 63);
  size_t b   = i / ((size_t)64 * HWSZ);
  size_t j = (b * HWSZ + pix) * 64 + c;
  o[i] = (float)t[j] + (float)x[j];
}

// ---------------------------------------------------------------------------
// conv3x3 as implicit GEMM on v_wmma_f32_16x16x32_f16, NHWC activations.
// Workgroup: 8 waves -> 64 couts x 64 pixels; each wave owns one 16-cout
// block x 32 pixels = TWO independent accumulators (breaks the WMMA RAW
// chain, halves A-fragment traffic).
// LDS stage: [row(3)][x(66)][cin] filled with GLOBAL_LOAD_ASYNC_TO_LDS_B128
// (ASYNCcnt path, no VGPR round-trip); OOB halo chunks get zero ds_stores.
// B fragment (n = lane%16, K = (lane/16)*16 + e) = one 32B ds load.
// Fused: +bias (f32), leaky(0.2), +residual (f16), b128 output stores.
// ---------------------------------------------------------------------------
template <int CIN>
__global__ __launch_bounds__(256) void conv3x3_wmma_k(
    const half_t* __restrict__ in, const half_t* __restrict__ wfrag,
    const float* __restrict__ bias, const half_t* __restrict__ res,
    half_t* __restrict__ out, int leaky_flag) {
  constexpr int KB = CIN / 32;
  __shared__ alignas(16) half_t lds[3 * 66 * CIN];

  const int x0  = blockIdx.x * 64;   // W/64 = 4
  const int y   = blockIdx.y;        // H
  const int b   = blockIdx.z;        // B
  const int tid = threadIdx.x;

  // stage 3 input rows (x0-1 .. x0+64) with zero-padded halo, 16B chunks
  constexpr int NC = 3 * 66 * CIN / 8;
  for (int j = tid; j < NC; j += 256) {
    int idx8 = j * 8;
    int c8 = idx8 % CIN;
    int p  = idx8 / CIN;
    int r  = p / 66;
    int xx = p - r * 66;
    int gy = y + r - 1;
    int gx = x0 + xx - 1;
    if ((unsigned)gy < HH && (unsigned)gx < WW) {
      unsigned ldsoff = (unsigned)(uintptr_t)&lds[idx8];
      unsigned goff = (unsigned)(((((size_t)b * HH + gy) * WW + gx) * CIN + c8) * 2);
      asm volatile("global_load_async_to_lds_b128 %0, %1, %2"
                   :: "v"(ldsoff), "v"(goff), "s"(in) : "memory");
    } else {
      *(uint4*)&lds[idx8] = (uint4){0u, 0u, 0u, 0u};
    }
  }
  asm volatile("s_wait_asynccnt 0x0" ::: "memory");
  __syncthreads();

  const int wave = tid >> 5;
  const int lane = tid & 31;
  const int mb = wave & 3;    // cout block (16)
  const int nh = wave >> 2;   // pixel half (32)
  const int n  = lane & 15;
  const int hi = lane >> 4;

  v8f acc0 = {}, acc1 = {};
  const int cb = mb * 16 + hi * 8;   // D layout: elem r -> M = r + hi*8
  if (bias) {
#pragma unroll
    for (int r = 0; r < 8; ++r) { acc0[r] = bias[cb + r]; acc1[r] = acc0[r]; }
  }

#pragma unroll
  for (int t = 0; t < 9; ++t) {
    const int dy = t / 3, dx = t % 3;
#pragma unroll
    for (int kb = 0; kb < KB; ++kb) {
      v16h a = *(const v16h*)(wfrag + ((((size_t)t * KB + kb) * 4 + mb) * 32 + lane) * 16);
      const int xb = nh * 32 + n + dx;
      v16h b0 = *(const v16h*)(&lds[(dy * 66 + xb) * CIN + kb * 32 + hi * 16]);
      v16h b1 = *(const v16h*)(&lds[(dy * 66 + xb + 16) * CIN + kb * 32 + hi * 16]);
      acc0 = __builtin_amdgcn_wmma_f32_16x16x32_f16(false, a, false, b0, (short)0, acc0, false, false);
      acc1 = __builtin_amdgcn_wmma_f32_16x16x32_f16(false, a, false, b1, (short)0, acc1, false, false);
    }
  }

#pragma unroll
  for (int s = 0; s < 2; ++s) {
    const v8f& acc = s ? acc1 : acc0;
    const int px = x0 + nh * 32 + s * 16 + n;
    size_t idx = (((size_t)b * HH + y) * WW + px) * COUT + cb;   // 8 contiguous ch
    v8h hv;
    if (res) {
      v8h rv = *(const v8h*)&res[idx];
#pragma unroll
      for (int r = 0; r < 8; ++r) hv[r] = (half_t)(acc[r] + (float)rv[r]);
    } else if (leaky_flag) {
#pragma unroll
      for (int r = 0; r < 8; ++r) {
        float v = acc[r];
        hv[r] = (half_t)(v > 0.f ? v : 0.2f * v);
      }
    } else {
#pragma unroll
      for (int r = 0; r < 8; ++r) hv[r] = (half_t)acc[r];
    }
    *(v8h*)&out[idx] = hv;
  }
}

// ---------------------------------------------------------------------------
// local_similar (NHWC): att[p][o=di*9+dj] = sum_c q[p][c] * k[p+(di-4,dj-4)][c]
// 8x16 pixel tile, 128 threads. k tile (16x24x64 f16 = 48KB) staged in LDS
// via async-to-LDS b128 (zero ds_stores on the halo); q in 32 half2 regs
// (8 b128 loads); inner loop v_dot2_f32_f16. att padded to 96 ch (zeros).
// ---------------------------------------------------------------------------
__global__ __launch_bounds__(128) void local_sim_k(
    const half_t* __restrict__ q, const half_t* __restrict__ k,
    half_t* __restrict__ att) {
  constexpr int C = 64;
  __shared__ alignas(16) half_t kt[16 * 24 * C];   // [yy][xx][c]

  const int x0 = blockIdx.x * 16;   // W/16
  const int y0 = blockIdx.y * 8;    // H/8
  const int b  = blockIdx.z;
  const int tid = threadIdx.x;

  constexpr int NC = 16 * 24 * C / 8;
  for (int j = tid; j < NC; j += 128) {
    int idx8 = j * 8;
    int c8 = idx8 & 63;
    int p  = idx8 >> 6;
    int yy = p / 24;
    int xx = p - yy * 24;
    int gy = y0 + yy - 4, gx = x0 + xx - 4;
    if ((unsigned)gy < HH && (unsigned)gx < WW) {
      unsigned ldsoff = (unsigned)(uintptr_t)&kt[idx8];
      unsigned goff = (unsigned)(((((size_t)b * HH + gy) * WW + gx) * C + c8) * 2);
      asm volatile("global_load_async_to_lds_b128 %0, %1, %2"
                   :: "v"(ldsoff), "v"(goff), "s"(k) : "memory");
    } else {
      *(uint4*)&kt[idx8] = (uint4){0u, 0u, 0u, 0u};
    }
  }
  asm volatile("s_wait_asynccnt 0x0" ::: "memory");
  __syncthreads();

  const int px = tid & 15;
  const int py = tid >> 4;
  const int gx = x0 + px, gy = y0 + py;

  h2v qr[32];
  {
    const half_t* qb = &q[(((size_t)b * HH + gy) * WW + gx) * C];
    uint4* qv = (uint4*)qr;
#pragma unroll
    for (int i = 0; i < 8; ++i) qv[i] = *(const uint4*)(qb + i * 8);
  }

  half_t* ab = &att[(((size_t)b * HH + gy) * WW + gx) * 96];
  for (int di = 0; di < 9; ++di) {
    for (int dj = 0; dj < 9; ++dj) {
      const half_t* kp = &kt[((py + di) * 24 + (px + dj)) * C];
      float acc = 0.f;
#pragma unroll
      for (int c2 = 0; c2 < 32; ++c2) {
#if __has_builtin(__builtin_amdgcn_fdot2)
        acc = __builtin_amdgcn_fdot2(qr[c2], *(const h2v*)(kp + 2 * c2), acc, false);
#else
        acc += (float)qr[c2][0] * (float)kp[2 * c2] +
               (float)qr[c2][1] * (float)kp[2 * c2 + 1];
#endif
      }
      ab[di * 9 + dj] = (half_t)acc;
    }
  }
  // zero the pad channels 81..95 (workspace is poisoned; conv<96> reads them)
#pragma unroll
  for (int o = 81; o < 96; ++o) ab[o] = (half_t)0;
}

// ---------------------------------------------------------------------------
extern "C" void kernel_launch(void* const* d_in, const int* in_sizes, int n_in,
                              void* d_out, int out_size, void* d_ws, size_t ws_size,
                              hipStream_t stream) {
  (void)in_sizes; (void)n_in; (void)out_size; (void)ws_size;
  const float* old_  = (const float*)d_in[0];
  const float* new_  = (const float*)d_in[1];
  const float* wq    = (const float*)d_in[2];
  const float* bq    = (const float*)d_in[3];
  const float* wk    = (const float*)d_in[4];
  const float* bk    = (const float*)d_in[5];
  const float* wv    = (const float*)d_in[6];
  const float* bv    = (const float*)d_in[7];
  const float* wo1   = (const float*)d_in[8];
  const float* bo1   = (const float*)d_in[9];
  const float* rw1   = (const float*)d_in[10];
  const float* rw2   = (const float*)d_in[11];
  const float* wo2   = (const float*)d_in[12];
  const float* bo2   = (const float*)d_in[13];

  char* ws = (char*)d_ws;
  const size_t T16 = (size_t)BB * 64 * HWSZ * 2;   // 32 MB f16 tensor
  half_t* x16  = (half_t*)(ws);
  half_t* bufA = (half_t*)(ws + T16);
  half_t* bufB = (half_t*)(ws + 2 * T16);
  half_t* attb = (half_t*)(ws + 3 * T16);          // 48 MB (96 ch), reused as bufC
  half_t* bufC = attb;
  char* wfbase = ws + 3 * T16 + (size_t)BB * 96 * HWSZ * 2;
  const size_t WF64 = 9 * 2 * 4 * 32 * 16 * 2;     // 73728 B per 64-cin frag
  const size_t WF96 = 9 * 3 * 4 * 32 * 16 * 2;     // 110592 B
  half_t* wq_f  = (half_t*)(wfbase);
  half_t* wk_f  = (half_t*)(wfbase + WF64);
  half_t* wo1_f = (half_t*)(wfbase + 2 * WF64);
  half_t* wo2_f = (half_t*)(wfbase + 3 * WF64);
  half_t* wv_f  = (half_t*)(wfbase + 4 * WF64);
  half_t* r1_f  = (half_t*)(wfbase + 4 * WF64 + WF96);
  half_t* r2_f  = (half_t*)(wfbase + 4 * WF64 + WF96 + 8 * WF64);

  // ---- pack weights into WMMA A-fragment layout
  {
    int tot64 = 9 * 2 * 4 * 32 * 16, g64 = (tot64 + 255) / 256;
    int tot96 = 9 * 3 * 4 * 32 * 16, g96 = (tot96 + 255) / 256;
    pack_wfrag_k<<<g64, 256, 0, stream>>>(wq,  wq_f,  64, 2);
    pack_wfrag_k<<<g64, 256, 0, stream>>>(wk,  wk_f,  64, 2);
    pack_wfrag_k<<<g64, 256, 0, stream>>>(wo1, wo1_f, 64, 2);
    pack_wfrag_k<<<g64, 256, 0, stream>>>(wo2, wo2_f, 64, 2);
    pack_wfrag_k<<<g96, 256, 0, stream>>>(wv,  wv_f,  81, 3);
    for (int i = 0; i < 8; ++i) {
      pack_wfrag_k<<<g64, 256, 0, stream>>>(rw1 + (size_t)i * 64 * 64 * 9,
                                            r1_f + i * (WF64 / 2), 64, 2);
      pack_wfrag_k<<<g64, 256, 0, stream>>>(rw2 + (size_t)i * 64 * 64 * 9,
                                            r2_f + i * (WF64 / 2), 64, 2);
    }
  }

  // ---- x = concat(old,new) -> f16 NHWC
  pack_x_k<<<(BB * 64 * HWSZ) / 256, 256, 0, stream>>>(old_, new_, x16);

  dim3 cgrid(WW / 64, HH, BB), cblk(256);
  // q, k
  conv3x3_wmma_k<64><<<cgrid, cblk, 0, stream>>>(x16, wq_f, bq, nullptr, bufA, 0);
  conv3x3_wmma_k<64><<<cgrid, cblk, 0, stream>>>(x16, wk_f, bk, nullptr, bufB, 0);
  // att = local_similar(q,k)  [96-channel padded NHWC]
  local_sim_k<<<dim3(WW / 16, HH / 8, BB), 128, 0, stream>>>(bufA, bufB, attb);
  // v = conv(att, wv)+bv
  conv3x3_wmma_k<96><<<cgrid, cblk, 0, stream>>>(attb, wv_f, bv, nullptr, bufA, 0);
  // xv = x * v
  mul_k<<<(BB * 64 * HWSZ) / 256, 256, 0, stream>>>(x16, bufA, bufB);
  // y = conv(xv, wo1)+bo1
  conv3x3_wmma_k<64><<<cgrid, cblk, 0, stream>>>(bufB, wo1_f, bo1, nullptr, bufA, 0);

  // 8 ResBlocks: y = y + conv(leaky(conv(y,w1)), w2)   (y ping-pongs A <-> C)
  half_t* y = bufA;
  half_t* yAlt = bufC;
  for (int i = 0; i < 8; ++i) {
    conv3x3_wmma_k<64><<<cgrid, cblk, 0, stream>>>(
        y, r1_f + i * (WF64 / 2), nullptr, nullptr, bufB, 1);
    conv3x3_wmma_k<64><<<cgrid, cblk, 0, stream>>>(
        bufB, r2_f + i * (WF64 / 2), nullptr, y, yAlt, 0);
    half_t* tswap = y; y = yAlt; yAlt = tswap;
  }

  // t = conv(y, wo2)+bo2  -> yAlt ;  out = t + x  (f32 NCHW)
  conv3x3_wmma_k<64><<<cgrid, cblk, 0, stream>>>(y, wo2_f, bo2, nullptr, yAlt, 0);
  final_k<<<(BB * 64 * HWSZ) / 256, 256, 0, stream>>>(yAlt, x16, (float*)d_out);
}
